// BasicMTL_31086973289277
// MI455X (gfx1250) — compile-verified, matrix operand order checked by reading, not run on hardware
//
#include <hip/hip_runtime.h>
#include <hip/hip_bf16.h>
#include <stdint.h>

// ---------------- problem constants (fixed by setup_inputs) ----------------
#define T_TOK   4096
#define HID     768
#define META    20
#define MAXW    30
#define UHDIM   1024
#define SPAN_DIM 2324          // 3*HID + META
#define KPAD    2336           // SPAN_DIM padded to 73*32 for K of WMMA
#define KT      73             // KPAD / 32
#define NSPAN   8192
#define KTOP    1024           // num_top_mentions
#define ATOP    50             // num_top_antecedents
#define FA_NPAD 2368           // SPAN_DIM padded to 37*64 for N tiling

typedef __attribute__((ext_vector_type(16))) __bf16 bf16x16;
typedef __attribute__((ext_vector_type(8)))  float  floatx8;
typedef __attribute__((ext_vector_type(4)))  unsigned int uint32x4;
typedef __attribute__((ext_vector_type(8)))  int int32x8;
typedef __attribute__((ext_vector_type(4)))  int int32x4;

#define NEG_INF (-__builtin_huge_valf())

#if __has_builtin(__builtin_amdgcn_tensor_load_to_lds) && \
    __has_builtin(__builtin_amdgcn_s_wait_tensorcnt)
#define USE_TDM 1
#else
#define USE_TDM 0
#endif

__device__ __forceinline__ unsigned short f2bf(float x) {
  unsigned int u = __float_as_uint(x);
  u += 0x7FFFu + ((u >> 16) & 1u);          // round-to-nearest-even
  return (unsigned short)(u >> 16);
}

__device__ __forceinline__ float wave_sum(float v) {
  for (int m = 16; m > 0; m >>= 1) v += __shfl_xor(v, m, 32);
  return v;
}
__device__ __forceinline__ float wave_max(float v) {
  for (int m = 16; m > 0; m >>= 1) { float o = __shfl_xor(v, m, 32); v = o > v ? o : v; }
  return v;
}

#if USE_TDM
// Issue one TDM 2D tile load: 64 rows x 32 bf16 cols from a (mtot x KPAD)
// bf16 matrix into LDS at lds_addr.  D# layout per CDNA5 ISA ch.8.
// 6-arg builtin form (clang-23 / therock-10.0): (g0, g1, g2, g3, g4, cpol).
__device__ __forceinline__ void tdm_load_tile64x32(const unsigned short* gsrc,
                                                   unsigned lds_addr, unsigned mtot) {
  unsigned long long ga = (unsigned long long)(uintptr_t)gsrc;
  uint32x4 g0 = { 1u,                                    // count=1, user desc
                  lds_addr,                              // LDS byte address
                  (unsigned)(ga & 0xffffffffu),          // global_addr[31:0]
                  ((unsigned)((ga >> 32) & 0x01ffffffu)) // global_addr[56:32]
                      | 0x80000000u };                   // type=2 ("image")
  int32x8 g1 = { (int)0x00010000u,                   // wg_mask=0, data_size=1 (2B)
                 (int)((KPAD & 0xffffu) << 16),      // tensor_dim0[15:0]
                 (int)((mtot & 0xffffu) << 16),      // dim0[31:16]=0 | tensor_dim1[15:0]
                 (int)((32u << 16) | (mtot >> 16)),  // tensor_dim1[31:16] | tile_dim0=32
                 (int)64,                            // tile_dim1=64, tile_dim2=0
                 (int)KPAD,                          // tensor_dim0_stride[31:0]
                 0, 0 };                             // stride hi / dim1_stride = 0
  int32x4 z4 = {0, 0, 0, 0};
  int32x8 z8 = {0, 0, 0, 0, 0, 0, 0, 0};
  __builtin_amdgcn_tensor_load_to_lds(g0, g1, z4, z4, z8, 0);
}
#endif

// ---------------- 1. token attention logits: tok_attn = H w + b -----------
__global__ void k_tok_attn(const float* __restrict__ hs, const float* __restrict__ w,
                           const float* __restrict__ b, float* __restrict__ tok_attn) {
  int wave = threadIdx.x >> 5, lane = threadIdx.x & 31;
  int t = blockIdx.x * 8 + wave;
  float acc = 0.f;
  for (int d = lane; d < HID; d += 32) acc += hs[(size_t)t * HID + d] * w[d];
  acc = wave_sum(acc);
  if (lane == 0) tok_attn[t] = acc + b[0];
}

// ---------------- 2. pack a K x N f32 weight into bf16 WMMA-B fragments ----
__global__ void k_pack_b(const float* __restrict__ W, unsigned short* __restrict__ Bp,
                         int N, int Npad) {
  int idx = blockIdx.x * blockDim.x + threadIdx.x;
  int total = KT * Npad * 2;
  if (idx >= total) return;
  int h  = idx & 1;
  int n  = (idx >> 1) % Npad;
  int kt = (idx >> 1) / Npad;
  size_t base = ((size_t)kt * Npad + n) * 32 + (size_t)h * 16;
  for (int j = 0; j < 16; ++j) {
    int k = kt * 32 + h * 16 + j;
    float v = (k < SPAN_DIM && n < N) ? W[(size_t)k * N + n] : 0.f;
    Bp[base + j] = f2bf(v);
  }
}

// ---------------- 3. span embeddings (windowed softmax attention) ----------
__global__ void k_span_emb(const float* __restrict__ hs, const int* __restrict__ starts,
                           const int* __restrict__ widths, const float* __restrict__ wemb,
                           const float* __restrict__ tok_attn,
                           float* __restrict__ emb, unsigned short* __restrict__ Abf) {
  int c = blockIdx.x;
  int tid = threadIdx.x;
  int s = starts[c], w = widths[c];
  int e = s + w;
  __shared__ float wt[32];
  if (tid < 32) {
    int lane = tid;
    float tv = (lane <= w) ? tok_attn[s + lane] : NEG_INF;
    float m = wave_max(tv);
    float p = (lane <= w) ? __expf(tv - m) : 0.f;
    float den = wave_sum(p);
    wt[lane] = p / den;
  }
  __syncthreads();
  float* er = emb + (size_t)c * SPAN_DIM;
  unsigned short* ar = Abf + (size_t)c * KPAD;
  for (int d = tid; d < HID; d += blockDim.x) {
    float vs = hs[(size_t)s * HID + d];
    float ve = hs[(size_t)e * HID + d];
    float acc = 0.f;
    for (int j = 0; j <= w; ++j) acc += wt[j] * hs[(size_t)(s + j) * HID + d];
    er[d] = vs; er[HID + d] = ve; er[2 * HID + META + d] = acc;
    ar[d] = f2bf(vs); ar[HID + d] = f2bf(ve); ar[2 * HID + META + d] = f2bf(acc);
  }
  if (tid < META) {
    float v = wemb[w * META + tid];
    er[2 * HID + tid] = v;
    ar[2 * HID + tid] = f2bf(v);
  }
  if (tid < KPAD - SPAN_DIM) ar[SPAN_DIM + tid] = 0;   // zero K padding
}

// ------- shared GEMM machinery: 64x64 block tile, 4 waves, 4 WMMA/wave/kt --
// A tile (64 x 32 bf16) is staged in LDS (TDM when available), each wave owns
// one 16-wide N tile and accumulates 4 M-subtiles.
#define GEMM_PROLOGUE()                                                        \
  const int m0 = blockIdx.x * 64;                                              \
  const int wave = threadIdx.x >> 5, lane = threadIdx.x & 31;                  \
  const int nT = blockIdx.y * 64 + wave * 16;                                  \
  __shared__ __align__(128) unsigned short As[64 * 32];                        \
  floatx8 c0 = {0,0,0,0,0,0,0,0}, c1 = {0,0,0,0,0,0,0,0};                      \
  floatx8 c2 = {0,0,0,0,0,0,0,0}, c3 = {0,0,0,0,0,0,0,0};                      \
  const unsigned ldsAs = (unsigned)(uintptr_t)(void*)&As[0];                   \
  (void)ldsAs;

#if USE_TDM
#define GEMM_STAGE_A(Aptr, MTOT)                                               \
  if (threadIdx.x < 32) {                                                      \
    tdm_load_tile64x32((Aptr) + (size_t)m0 * KPAD + kt * 32, ldsAs, (MTOT));   \
    __builtin_amdgcn_s_wait_tensorcnt(0);                                      \
  }                                                                            \
  __syncthreads();
#else
#define GEMM_STAGE_A(Aptr, MTOT)                                               \
  {                                                                            \
    for (int p = 0; p < 4; ++p) {                                              \
      int ch = threadIdx.x + p * 128;   /* 512 chunks of 4 bf16 */             \
      int rw = ch >> 3, cl = (ch & 7) * 4;                                     \
      *(unsigned long long*)&As[rw * 32 + cl] =                                \
          *(const unsigned long long*)&(Aptr)[(size_t)(m0 + rw) * KPAD +       \
                                              kt * 32 + cl];                   \
    }                                                                          \
  }                                                                            \
  __syncthreads();
#endif

#define GEMM_MACS()                                                            \
  {                                                                            \
    bf16x16 a0 = *(const bf16x16*)&As[(0 * 16 + (lane & 15)) * 32 + (lane >> 4) * 16]; \
    bf16x16 a1 = *(const bf16x16*)&As[(1 * 16 + (lane & 15)) * 32 + (lane >> 4) * 16]; \
    bf16x16 a2 = *(const bf16x16*)&As[(2 * 16 + (lane & 15)) * 32 + (lane >> 4) * 16]; \
    bf16x16 a3 = *(const bf16x16*)&As[(3 * 16 + (lane & 15)) * 32 + (lane >> 4) * 16]; \
    c0 = __builtin_amdgcn_wmma_f32_16x16x32_bf16(false, a0, false, b, (short)0, c0, false, false); \
    c1 = __builtin_amdgcn_wmma_f32_16x16x32_bf16(false, a1, false, b, (short)0, c1, false, false); \
    c2 = __builtin_amdgcn_wmma_f32_16x16x32_bf16(false, a2, false, b, (short)0, c2, false, false); \
    c3 = __builtin_amdgcn_wmma_f32_16x16x32_bf16(false, a3, false, b, (short)0, c3, false, false); \
  }                                                                            \
  __syncthreads();

// ---------------- 4. WMMA GEMM: H = relu(span_emb @ u_w1 + b1) -------------
__global__ void k_gemm_mention(const unsigned short* __restrict__ A,
                               const unsigned short* __restrict__ Bp,
                               const float* __restrict__ b1, float* __restrict__ H) {
  GEMM_PROLOGUE();
  for (int kt = 0; kt < KT; ++kt) {
    GEMM_STAGE_A(A, NSPAN);
    bf16x16 b = *(const bf16x16*)(Bp + ((size_t)kt * UHDIM + (nT + (lane & 15))) * 32 +
                                  (lane >> 4) * 16);
    if (kt + 1 < KT)
      __builtin_prefetch(Bp + ((size_t)(kt + 1) * UHDIM + (nT + (lane & 15))) * 32, 0, 1);
    GEMM_MACS();
  }
  const int n = nT + (lane & 15);
  const int mb = m0 + ((lane >> 4) << 3);
  floatx8 cc[4] = {c0, c1, c2, c3};
#pragma unroll
  for (int t = 0; t < 4; ++t)
#pragma unroll
    for (int r = 0; r < 8; ++r) {
      float v = cc[t][r] + b1[n];
      H[(size_t)(mb + t * 16 + r) * UHDIM + n] = v > 0.f ? v : 0.f;
    }
}

// ---------------- 5. mention score GEMV: m = H @ u_w2 + b2 -----------------
__global__ void k_mention_reduce(const float* __restrict__ H, const float* __restrict__ w2,
                                 const float* __restrict__ b2, float* __restrict__ mention) {
  int wave = threadIdx.x >> 5, lane = threadIdx.x & 31;
  int r = blockIdx.x * 8 + wave;
  float acc = 0.f;
  for (int n = lane; n < UHDIM; n += 32) acc += H[(size_t)r * UHDIM + n] * w2[n];
  acc = wave_sum(acc);
  if (lane == 0) mention[r] = acc + b2[0];
}

// ---------------- 6. serial greedy non-crossing selection ------------------
__global__ void __launch_bounds__(1024)
k_select(const float* __restrict__ mention, const int* __restrict__ starts,
         const int* __restrict__ widths, int* s2e, int* e2s, int* sel) {
  __shared__ unsigned long long keys[NSPAN];   // 64 KB
  int tid = threadIdx.x;
  for (int i = tid; i < NSPAN; i += 1024) {
    unsigned int u = __float_as_uint(mention[i]);
    u = (u & 0x80000000u) ? ~u : (u | 0x80000000u);   // ascending-orderable
    unsigned int km = ~u;                              // descending by score
    keys[i] = ((unsigned long long)km << 32) | (unsigned int)i;
  }
  for (int i = tid; i < T_TOK; i += 1024) { s2e[i] = -1; e2s[i] = T_TOK + 1; }
  for (int i = tid; i < KTOP; i += 1024) sel[i] = NSPAN - 1;
  __syncthreads();
  for (int size = 2; size <= NSPAN; size <<= 1)
    for (int stride = size >> 1; stride > 0; stride >>= 1) {
      for (int t = tid; t < NSPAN / 2; t += 1024) {
        int i = 2 * t - (t & (stride - 1));
        int j = i + stride;
        bool up = ((i & size) == 0);
        unsigned long long a = keys[i], b = keys[j];
        if ((a > b) == up) { keys[i] = b; keys[j] = a; }
      }
      __syncthreads();
    }
  if (tid < 32) {
    volatile int* vs2e = s2e;
    volatile int* ve2s = e2s;
    int lane = tid;
    int cnt = 0;
    for (int it = 0; it < NSPAN && cnt < KTOP; ++it) {
      int ind = (int)(keys[it] & 0xFFFFFFFFull);
      int cs = starts[ind];
      int ce = cs + widths[ind];
      int t1 = cs + 1 + lane;
      int t2 = cs + lane;
      bool c1 = (t1 <= ce) && (vs2e[t1] > ce);
      bool c2 = (t2 < ce) && (ve2s[t2] < cs);
      bool crossing = __any(c1 || c2) != 0;
      if (!crossing) {
        if (lane == 0) {
          sel[cnt] = ind;
          if (ce > vs2e[cs]) vs2e[cs] = ce;
          if (cs < ve2s[ce]) ve2s[ce] = cs;
        }
        cnt++;
      }
      __threadfence_block();
    }
  }
  __syncthreads();
  for (int i = tid; i < KTOP; i += 1024) keys[i] = (unsigned long long)(unsigned int)sel[i];
  __syncthreads();
  for (int size = 2; size <= KTOP; size <<= 1)
    for (int stride = size >> 1; stride > 0; stride >>= 1) {
      if (tid < KTOP / 2) {
        int i = 2 * tid - (tid & (stride - 1));
        int j = i + stride;
        bool up = ((i & size) == 0);
        unsigned long long a = keys[i], b = keys[j];
        if ((a > b) == up) { keys[i] = b; keys[j] = a; }
      }
      __syncthreads();
    }
  for (int i = tid; i < KTOP; i += 1024) sel[i] = (int)keys[i];
}

// ---------------- 7. gather top embeddings + scores ------------------------
__global__ void k_gather_top(const float* __restrict__ emb, const float* __restrict__ mention,
                             const int* __restrict__ sel, unsigned short* __restrict__ topA,
                             float* __restrict__ sc) {
  int k = blockIdx.x, tid = threadIdx.x;
  int ind = sel[k];
  if (tid == 0) sc[k] = mention[ind];
  const float* er = emb + (size_t)ind * SPAN_DIM;
  unsigned short* tr = topA + (size_t)k * KPAD;
  for (int d = tid; d < SPAN_DIM; d += blockDim.x) tr[d] = f2bf(er[d]);
  if (tid < KPAD - SPAN_DIM) tr[SPAN_DIM + tid] = 0;
}

// ---------------- 8. WMMA GEMM: P = topA @ fa_w + fa_b (bf16 out) ----------
__global__ void k_gemm_fa(const unsigned short* __restrict__ A,
                          const unsigned short* __restrict__ Bp,
                          const float* __restrict__ fb, unsigned short* __restrict__ P) {
  GEMM_PROLOGUE();
  for (int kt = 0; kt < KT; ++kt) {
    GEMM_STAGE_A(A, KTOP);
    bf16x16 b = *(const bf16x16*)(Bp + ((size_t)kt * FA_NPAD + (nT + (lane & 15))) * 32 +
                                  (lane >> 4) * 16);
    if (kt + 1 < KT)
      __builtin_prefetch(Bp + ((size_t)(kt + 1) * FA_NPAD + (nT + (lane & 15))) * 32, 0, 1);
    GEMM_MACS();
  }
  const int n = nT + (lane & 15);
  const int mb = m0 + ((lane >> 4) << 3);
  floatx8 cc[4] = {c0, c1, c2, c3};
#pragma unroll
  for (int t = 0; t < 4; ++t)
#pragma unroll
    for (int r = 0; r < 8; ++r) {
      if (n < KPAD) {
        float v = (n < SPAN_DIM) ? (cc[t][r] + fb[n]) : 0.f;
        P[(size_t)(mb + t * 16 + r) * KPAD + n] = f2bf(v);
      }
    }
}

// ---------------- 9. WMMA GEMM: fast = P @ topA^T + scores + mask ----------
__global__ void k_gemm_att(const unsigned short* __restrict__ A,   // P   (KTOP x KPAD)
                           const unsigned short* __restrict__ Bt,  // topA(KTOP x KPAD)
                           const float* __restrict__ sc, float* __restrict__ fast) {
  GEMM_PROLOGUE();
  for (int kt = 0; kt < KT; ++kt) {
    GEMM_STAGE_A(A, KTOP);
    // B = topA^T: column n of B is row n of topA -> contiguous along K
    bf16x16 b = *(const bf16x16*)(Bt + (size_t)(nT + (lane & 15)) * KPAD + kt * 32 +
                                  (lane >> 4) * 16);
    if (kt + 1 < KT)
      __builtin_prefetch(Bt + (size_t)(nT + (lane & 15)) * KPAD + (kt + 1) * 32, 0, 1);
    GEMM_MACS();
  }
  const int n = nT + (lane & 15);
  const int mb = m0 + ((lane >> 4) << 3);
  floatx8 cc[4] = {c0, c1, c2, c3};
#pragma unroll
  for (int t = 0; t < 4; ++t)
#pragma unroll
    for (int r = 0; r < 8; ++r) {
      int i = mb + t * 16 + r, j = n;
      fast[(size_t)i * KTOP + j] = (j < i) ? (cc[t][r] + sc[i] + sc[j]) : NEG_INF;
    }
}

// ---------------- 10. row-wise top-50 --------------------------------------
__global__ void k_topk(const float* __restrict__ fast, float* __restrict__ out) {
  int row = blockIdx.x, tid = threadIdx.x;
  __shared__ float v[KTOP];
  __shared__ float rv[256];
  __shared__ int ri[256];
  for (int j = tid; j < KTOP; j += 256) v[j] = fast[(size_t)row * KTOP + j];
  __syncthreads();
  for (int a = 0; a < ATOP; ++a) {
    float bv = NEG_INF; int bi = KTOP;
    for (int j = tid; j < KTOP; j += 256)
      if (v[j] > bv || (v[j] == bv && j < bi)) { bv = v[j]; bi = j; }
    rv[tid] = bv; ri[tid] = bi;
    __syncthreads();
    for (int s = 128; s > 0; s >>= 1) {
      if (tid < s) {
        float ov = rv[tid + s]; int oi = ri[tid + s];
        if (ov > rv[tid] || (ov == rv[tid] && oi < ri[tid])) { rv[tid] = ov; ri[tid] = oi; }
      }
      __syncthreads();
    }
    if (tid == 0) {
      out[(size_t)row * ATOP + a] = rv[0];
      if (ri[0] < KTOP) v[ri[0]] = NEG_INF;
    }
    __syncthreads();
  }
}

// ---------------- driver ---------------------------------------------------
extern "C" void kernel_launch(void* const* d_in, const int* in_sizes, int n_in,
                              void* d_out, int out_size, void* d_ws, size_t ws_size,
                              hipStream_t stream) {
  (void)in_sizes; (void)n_in; (void)out_size; (void)ws_size;
  const float* hs     = (const float*)d_in[0];
  const int*   starts = (const int*)d_in[1];
  const int*   widths = (const int*)d_in[2];
  const float* wemb   = (const float*)d_in[3];
  const float* w_attn = (const float*)d_in[4];
  const float* b_attn = (const float*)d_in[5];
  const float* u_w1   = (const float*)d_in[6];
  const float* u_b1   = (const float*)d_in[7];
  const float* u_w2   = (const float*)d_in[8];
  const float* u_b2   = (const float*)d_in[9];
  const float* fa_w   = (const float*)d_in[10];
  const float* fa_b   = (const float*)d_in[11];
  float* out = (float*)d_out;

  char* ws = (char*)d_ws;
  size_t cur = 0;
  auto carve = [&](size_t bytes) -> void* {
    void* p = (void*)(ws + cur);
    cur = (cur + bytes + 255) & ~(size_t)255;
    return p;
  };
  float*          tok_attn = (float*)carve((size_t)T_TOK * 4);
  float*          emb      = (float*)carve((size_t)NSPAN * SPAN_DIM * 4);
  unsigned short* Abf      = (unsigned short*)carve((size_t)NSPAN * KPAD * 2);
  unsigned short* BpU      = (unsigned short*)carve((size_t)KT * UHDIM * 32 * 2);
  unsigned short* BpF      = (unsigned short*)carve((size_t)KT * FA_NPAD * 32 * 2);
  float*          H        = (float*)carve((size_t)NSPAN * UHDIM * 4);
  float*          mention  = (float*)carve((size_t)NSPAN * 4);
  int*            s2e      = (int*)carve((size_t)T_TOK * 4);
  int*            e2s      = (int*)carve((size_t)T_TOK * 4);
  int*            sel      = (int*)carve((size_t)KTOP * 4);
  unsigned short* topA     = (unsigned short*)carve((size_t)KTOP * KPAD * 2);
  float*          sc       = (float*)carve((size_t)KTOP * 4);
  unsigned short* Pbf      = (unsigned short*)carve((size_t)KTOP * KPAD * 2);
  float*          fastm    = (float*)carve((size_t)KTOP * KTOP * 4);

  k_tok_attn<<<T_TOK / 8, 256, 0, stream>>>(hs, w_attn, b_attn, tok_attn);
  k_pack_b<<<(KT * UHDIM * 2 + 255) / 256, 256, 0, stream>>>(u_w1, BpU, UHDIM, UHDIM);
  k_pack_b<<<(KT * FA_NPAD * 2 + 255) / 256, 256, 0, stream>>>(fa_w, BpF, SPAN_DIM, FA_NPAD);
  k_span_emb<<<NSPAN, 128, 0, stream>>>(hs, starts, widths, wemb, tok_attn, emb, Abf);
  k_gemm_mention<<<dim3(NSPAN / 64, UHDIM / 64), 128, 0, stream>>>(Abf, BpU, u_b1, H);
  k_mention_reduce<<<NSPAN / 8, 256, 0, stream>>>(H, u_w2, u_b2, mention);
  k_select<<<1, 1024, 0, stream>>>(mention, starts, widths, s2e, e2s, sel);
  k_gather_top<<<KTOP, 256, 0, stream>>>(emb, mention, sel, topA, sc);
  k_gemm_fa<<<dim3(KTOP / 64, FA_NPAD / 64), 128, 0, stream>>>(topA, BpF, fa_b, Pbf);
  k_gemm_att<<<dim3(KTOP / 64, KTOP / 64), 128, 0, stream>>>(Pbf, topA, sc, fastm);
  k_topk<<<KTOP, 256, 0, stream>>>(fastm, out);
}
